// LayerConfigurator_84868553769163
// MI455X (gfx1250) — matrix operand order repaired and checked by reading, not compiled, not run on hardware
//
#include <hip/hip_runtime.h>

typedef __attribute__((ext_vector_type(16))) _Float16 v16h;
typedef __attribute__((ext_vector_type(8)))  float    v8f;

#define EPSF 1e-5f
#define DCONST 2048          // D fixed by the reference; lets row strides fold to immediates

// ---------------------------------------------------------------------------
// Phase 1: gate mask = (logit1 > logit0) where logits come from
// RMSNorm -> causal depthwise conv (W=4) -> SiLU -> proj[2,D].
// Block = 256 threads = 8 waves; each wave owns 16 consecutive tokens and
// accumulates its 16x16 logits tile with v_wmma_f32_16x16x32_f16 over D.
// ---------------------------------------------------------------------------
__global__ __launch_bounds__(256)
void gate_kernel(const float* __restrict__ x,
                 const float* __restrict__ norm_w,
                 const float* __restrict__ conv_w,
                 const float* __restrict__ proj_w,
                 int* __restrict__ maskArr,
                 int S)
{
    constexpr int D = DCONST;
    __shared__ float ldsInv[136];                          // inv_rms, tokens s0-3..s0+127
    __shared__ __align__(16) _Float16 ldsProj[2 * D + 32]; // f16 proj rows + 16-dword zero pad
    __shared__ __align__(16) _Float16 aStg[8][16 * 32];    // per-wave activation staging

    const int tilesPerBatch = S / 128;
    const int b    = blockIdx.x / tilesPerBatch;
    const int tile = blockIdx.x % tilesPerBatch;
    const int s0   = tile * 128;

    const int tid  = threadIdx.x;
    const int lane = tid & 31;
    const int wid  = tid >> 5;

    // Stage the 2 projection rows as f16, plus a zeroed pad region used as the
    // B-matrix source for padding lanes (N >= 2).
    for (int i = tid; i < 2 * D; i += 256)
        ldsProj[i] = (_Float16)proj_w[i];
    if (tid < 32) ldsProj[2 * D + tid] = (_Float16)0.f;

    // inv_rms for the 128 tokens + 3-token causal halo (131 rows).
    // Padded rows (s<0) get inv=0, which arithmetically kills their contribution.
    for (int t = wid; t < 131; t += 8) {
        int s = s0 - 3 + t;
        float inv = 0.f;
        if (s >= 0) {
            const float* row = x + ((size_t)b * S + s) * D;
            float acc = 0.f;
            for (int d = lane; d < D; d += 32) { float v = row[d]; acc += v * v; }
            #pragma unroll
            for (int off = 16; off > 0; off >>= 1)
                acc += __shfl_xor(acc, off, 32);
            inv = rsqrtf(acc / (float)D + EPSF);
        }
        if (lane == 0) ldsInv[t] = inv;
    }
    __syncthreads();

    const int s0w = s0 + wid * 16;     // first token of this wave's 16-token tile
    const int ll  = lane & 15;
    const int hi  = lane >> 4;

    // Chunk-invariant state. Row bases are wave-uniform (depend only on b/tile/wid/r):
    // force them into SGPRs via readfirstlane so loads become saddr-form with a
    // single shared VGPR offset instead of 19 VGPR 64-bit pointers.
    float inv[19];
    const float* rbase[19];
    #pragma unroll
    for (int r = 0; r < 19; ++r) {
        inv[r] = ldsInv[wid * 16 + r];                     // 0 for padded rows
        int s  = s0w - 3 + r;
        int sc = (s < 0) ? 0 : s;
        int ro = __builtin_amdgcn_readfirstlane(b * S + sc);  // uniform row index
        rbase[r] = x + (size_t)ro * D;                     // scalar (SGPR) base
    }

    v8f c = {};                                            // 16x16 f32 logits accumulator

    const unsigned int* projU = (const unsigned int*)ldsProj;
    const unsigned int* aU    = (const unsigned int*)(&aStg[wid][0]);

    // B gather source: proj row ll for lanes N<2, zeroed pad (stationary) otherwise.
    const unsigned int* bPtr = projU + ((ll < 2) ? ll * (D >> 1) : D) + (hi << 2);
    const int bStep = (ll < 2) ? 16 : 0;                   // dwords per chunk

    int dl = lane;                                         // shared load offset (elements)
    const int KC = D / 32;
    for (int kc = 0; kc < KC; ++kc) {
        // Fold norm weight into the 4 conv taps (conv is linear in hn).
        float  nw = norm_w[dl];
        float4 cw = ((const float4*)conv_w)[dl];
        float c0 = cw.x * nw, c1 = cw.y * nw, c2 = cw.z * nw, c3 = cw.w * nw;

        // 19 halo rows for this lane's d-column (coalesced, saddr-form, branch-free).
        float hn[19];
        #pragma unroll
        for (int r = 0; r < 19; ++r)
            hn[r] = rbase[r][dl] * inv[r];

        __builtin_prefetch(rbase[9] + dl + 32, 0, 3);      // global_prefetch_b8, near

        // Causal depthwise conv + SiLU (fast rcp; only the sign of the logit
        // difference matters downstream); stage f16 (row = token, col = K = lane).
        #pragma unroll
        for (int s = 0; s < 16; ++s) {
            float y = hn[s] * c0 + hn[s + 1] * c1 + hn[s + 2] * c2 + hn[s + 3] * c3;
            float a = y * __builtin_amdgcn_rcpf(1.f + __expf(-y));
            aStg[wid][s * 32 + lane] = (_Float16)a;
        }
        __builtin_amdgcn_wave_barrier();   // intra-wave LDS store->load ordering

        // A gather per 16-bit 16x32 A layout: lane = M (mod 16), pair index
        // p = (v&3) + (v>>2)*8 + hi*4  ->  two contiguous b128 reads.
        union Pack { uint4 q[2]; v16h h; } A, B;
        const uint4* a4 = (const uint4*)(aU + ll * 16 + (hi << 2));
        A.q[0] = a4[0];                    // p = hi*4 + {0..3}
        A.q[1] = a4[2];                    // p = 8 + hi*4 + {0..3}

        // B gather: lane = N, same pair index p = (v&3) + (v>>2)*8 + hi*4 as A.
        const uint4* b4 = (const uint4*)bPtr;
        B.q[0] = b4[0];
        B.q[1] = b4[2];
        bPtr += bStep;
        __builtin_amdgcn_wave_barrier();   // reads done before next iter's stores

        c = __builtin_amdgcn_wmma_f32_16x16x32_f16(false, A.h, false, B.h,
                                                   (short)0, c, false, false);
        dl += 32;
    }

    // C/D layout: lane l, reg r -> row M = r + 8*(l>=16), col N = l%16.
    // logit0/logit1 of the same token sit in adjacent lanes (N=0 / N=1), so a
    // single shfl_xor(1) pairs them up. Lanes 0 and 16 then own the decisions
    // for tokens 0..7 / 8..15 and emit them as two coalesced b128 stores.
    int mk[8];
    #pragma unroll
    for (int i = 0; i < 8; ++i) {
        float mine  = c[i];
        float other = __shfl_xor(mine, 1, 32);
        // even lane (N=0): mask = logit1 > logit0 = other > mine
        mk[i] = (other > mine) ? 1 : 0;
    }
    if (ll == 0) {                               // lanes 0 and 16
        int4* p = (int4*)(maskArr + (size_t)b * S + s0w + (hi << 3));
        p[0] = make_int4(mk[0], mk[1], mk[2], mk[3]);
        p[1] = make_int4(mk[4], mk[5], mk[6], mk[7]);
    }
}

// ---------------------------------------------------------------------------
// Phase 2: per-batch inclusive scan of mask; emit inverse map srcOf[dest]=s
// and per-batch selected count. One 1024-thread block per batch.
// ---------------------------------------------------------------------------
__global__ __launch_bounds__(1024)
void scan_kernel(const int* __restrict__ maskArr,
                 int* __restrict__ srcOf,
                 int* __restrict__ nsel,
                 int S)
{
    const int b    = blockIdx.x;
    const int t    = threadIdx.x;
    const int lane = t & 31;
    const int wv   = t >> 5;
    const int PT   = S / 1024;          // 4 for S=4096 (<= 8 supported)
    __shared__ int wsum[32];

    int m[8];
    int tot = 0;
    const int base = t * PT;
    for (int i = 0; i < PT; ++i) {
        m[i] = maskArr[(size_t)b * S + base + i];
        tot += m[i];
    }

    int sc = tot;
    #pragma unroll
    for (int off = 1; off < 32; off <<= 1) {
        int n = __shfl_up(sc, off, 32);
        if (lane >= off) sc += n;
    }
    if (lane == 31) wsum[wv] = sc;
    __syncthreads();
    if (wv == 0) {
        int v = wsum[lane];
        #pragma unroll
        for (int off = 1; off < 32; off <<= 1) {
            int n = __shfl_up(v, off, 32);
            if (lane >= off) v += n;
        }
        wsum[lane] = v;
    }
    __syncthreads();

    const int waveOff = (wv > 0) ? wsum[wv - 1] : 0;
    int run = waveOff + sc - tot;       // exclusive prefix for this thread
    for (int i = 0; i < PT; ++i) {
        run += m[i];
        if (m[i]) srcOf[(size_t)b * S + run - 1] = base + i;
    }
    if (t == 1023) nsel[b] = run;
}

// ---------------------------------------------------------------------------
// Phase 3: write output exactly once. Row r < nsel[b] -> copy x[srcOf[r]],
// else zeros. One 256-thread block per output row; float4 coalesced.
// ---------------------------------------------------------------------------
__global__ __launch_bounds__(256)
void scatter_kernel(const float* __restrict__ x,
                    const int* __restrict__ srcOf,
                    const int* __restrict__ nsel,
                    float* __restrict__ out,
                    int S)
{
    constexpr int D = DCONST;
    const int row = blockIdx.x;
    const int b   = row / S;
    const int r   = row % S;
    const int n   = nsel[b];
    const int nd4 = D >> 2;
    float4* dst = (float4*)(out + ((size_t)b * S + r) * D);
    if (r < n) {
        int s = srcOf[(size_t)b * S + r];
        const float4* src = (const float4*)(x + ((size_t)b * S + s) * D);
        for (int i = threadIdx.x; i < nd4; i += 256) dst[i] = src[i];
    } else {
        float4 z = {0.f, 0.f, 0.f, 0.f};
        for (int i = threadIdx.x; i < nd4; i += 256) dst[i] = z;
    }
}

// ---------------------------------------------------------------------------
extern "C" void kernel_launch(void* const* d_in, const int* in_sizes, int n_in,
                              void* d_out, int out_size, void* d_ws, size_t ws_size,
                              hipStream_t stream) {
    const float* x  = (const float*)d_in[0];   // [B,S,D]
    const float* nw = (const float*)d_in[1];   // [D]
    const float* cw = (const float*)d_in[2];   // [D,4]
    const float* pw = (const float*)d_in[3];   // [2,D]
    float* out = (float*)d_out;                // [B,S,D]

    const int D = DCONST;                      // 2048 (fixed by reference)
    const int B = 4;
    const int S = in_sizes[0] / (B * D);       // 4096

    int* maskArr = (int*)d_ws;                 // B*S ints
    int* srcOf   = maskArr + (size_t)B * S;    // B*S ints
    int* nsel    = srcOf   + (size_t)B * S;    // B ints

    gate_kernel<<<B * (S / 128), 256, 0, stream>>>(x, nw, cw, pw, maskArr, S);
    scan_kernel<<<B, 1024, 0, stream>>>(maskArr, srcOf, nsel, S);
    scatter_kernel<<<B * S, 256, 0, stream>>>(x, srcOf, nsel, out, S);
}
